// S5SSM_82592221102685
// MI455X (gfx1250) — compile-verified, matrix operand order checked by reading, not run on hardware
//
#include <hip/hip_runtime.h>

typedef __attribute__((ext_vector_type(16))) _Float16     v16h;
typedef __attribute__((ext_vector_type(8)))  _Float16     v8h;
typedef __attribute__((ext_vector_type(8)))  float        v8f;
typedef __attribute__((ext_vector_type(4)))  float        f32x4;
typedef __attribute__((ext_vector_type(4)))  unsigned int u32x4;
typedef __attribute__((ext_vector_type(8)))  int          i32x8;
typedef __attribute__((ext_vector_type(4)))  int          i32x4;

constexpr int BATCH  = 32;
constexpr int SEQ    = 4096;
constexpr int DMODEL = 256;
constexpr int NST    = 64;            // d_state
constexpr int TCH    = 64;            // chunk length
constexpr int NCH    = SEQ / TCH;     // 64 chunks

#if defined(__HIP_DEVICE_COMPILE__) && __has_builtin(__builtin_amdgcn_tensor_load_to_lds)
#define HAVE_TDM 1
#else
#define HAVE_TDM 0
#endif

// Load a 16x32 f16 WMMA operand tile from an LDS matrix stored with 64-half rows.
// Lane (m = lane&15, h = lane>>4) holds row m, K-segments [kb+8h, +8) and
// [kb+16+8h, +8) -> two ds_load_b128. B operands use the same addressing on a
// transposed (column-major) LDS image.
__device__ __forceinline__ v16h load_tile16(const _Float16* base, int row0, int kb, int lane) {
  const int m = lane & 15;
  const int h = lane >> 4;
  const _Float16* p = base + (row0 + m) * 64 + kb + 8 * h;
  union { v16h v; v8h q[2]; } r;
  r.q[0] = *(const v8h*)(p);
  r.q[1] = *(const v8h*)(p + 16);
  return r.v;
}

#if HAVE_TDM
// Issue one TDM load of a u-chunk tile: (1 x 64t x 32b) f32 elements at
// strides (1, DMODEL, SEQ*DMODEL), packed contiguously into LDS as [b][t].
__device__ __forceinline__ void tdm_load_chunk(const float* gptr, unsigned lds_off) {
  const unsigned long long ga = (unsigned long long)(uintptr_t)gptr;
  u32x4 g0;
  g0[0] = 1u;                                              // count=1 (valid), user mode
  g0[1] = lds_off;                                         // lds_addr (bytes)
  g0[2] = (unsigned)(ga & 0xffffffffu);                    // global_addr[31:0]
  g0[3] = (unsigned)((ga >> 32) & 0x1ffffffu) | (2u << 30); // global_addr[56:32] | type=2

  i32x8 g1;
  g1[0] = 0x00020000;                      // workgroup_mask=0, data_size=2 (4B)
  g1[1] = (int)(256u << 16);               // tensor_dim0[15:0] = DMODEL
  g1[2] = (int)(4096u << 16);              // tensor_dim0 hi=0 | tensor_dim1[15:0] = SEQ
  g1[3] = (int)(1u << 16);                 // tensor_dim1 hi=0 | tile_dim0 = 1
  g1[4] = (int)((32u << 16) | 64u);        // tile_dim1 = 64 (t) | tile_dim2 = 32 (b)
  g1[5] = 256;                             // tensor_dim0_stride = DMODEL
  g1[6] = 0;                               // dim0_stride hi | dim1_stride[15:0] (=0)
  g1[7] = (int)(1048576u >> 16);           // tensor_dim1_stride = SEQ*DMODEL -> [47:16]

  i32x4 g2;
  g2[0] = 32;                              // tensor_dim2 = BATCH
  g2[1] = 0; g2[2] = 0; g2[3] = 0;         // tensor_dim3 / dim2_stride / tile_dim3 unused
  i32x4 g3 = {0, 0, 0, 0};

#if __clang_major__ >= 23
  i32x8 g4 = {0, 0, 0, 0, 0, 0, 0, 0};
  __builtin_amdgcn_tensor_load_to_lds(g0, g1, g2, g3, g4, 0);
#else
  __builtin_amdgcn_tensor_load_to_lds(g0, g1, g2, g3, 0);
#endif
}
#endif

__global__ __launch_bounds__(256)
void s5_ssd_wmma(const float* __restrict__ u,
                 const float* __restrict__ log_dt,
                 const float* __restrict__ A_real,
                 const float* __restrict__ Bmat,
                 const float* __restrict__ Cmat,
                 const float* __restrict__ x0,
                 float* __restrict__ out) {
  // Per-channel constant matrices (f16, WMMA-operand layouts)
  __shared__ _Float16 Km[TCH * 64];   // K[t,s]  = k[t-s] (lower tri Toeplitz)
  __shared__ _Float16 Cp[TCH * 64];   // Cp[t,n] = C_n * At_n^(t+1)
  __shared__ _Float16 Bp[NST * 64];   // Bp[n,s] = Bt_n * At_n^(63-s)
  // Per-chunk staging
  __shared__ float    Uraw[BATCH * TCH]; // TDM destination: raw f32 u chunk [b][t]
  __shared__ _Float16 Ut[BATCH * 64];    // f16 image of u chunk (WMMA B-operand)
  __shared__ _Float16 Xt[BATCH * 64];    // f16 image of previous state
  __shared__ float Af[NST], Btf[NST], Cf[NST], decayf[NST], kvec[TCH];

  const int tid  = threadIdx.x;
  const int lane = tid & 31;
  const int wv   = tid >> 5;          // 8 wave32s
  const int d    = blockIdx.x;        // one channel per block

  const float dt = __expf(log_dt[d]);

  if (tid < NST) {
    const float a  = A_real[d * NST + tid];
    const float ad = a * dt;
    const float At = __expf(ad);
    Af[tid]     = ad;
    Btf[tid]    = (1.0f - At) * Bmat[d * NST + tid] / a;
    Cf[tid]     = Cmat[d * NST + tid];
    decayf[tid] = __expf(ad * (float)TCH);              // per-chunk decay At^T
  }
  __syncthreads();

  if (tid < TCH) {                                      // k[tau] = sum_n C*Bt*At^tau
    float s = 0.0f;
    for (int n = 0; n < NST; ++n)
      s += Cf[n] * Btf[n] * __expf(Af[n] * (float)tid);
    kvec[tid] = s;
  }
  __syncthreads();

  for (int e = tid; e < TCH * 64; e += 256) {
    const int r = e >> 6;
    const int c = e & 63;
    Km[e] = (r >= c) ? (_Float16)kvec[r - c] : (_Float16)(0.0f);
    Cp[e] = (_Float16)(Cf[c]  * __expf(Af[c] * (float)(r + 1)));
    Bp[e] = (_Float16)(Btf[r] * __expf(Af[r] * (float)(63 - c)));
  }
  __syncthreads();

  const int i   = wv >> 1;            // tile row (time rows for Y, state rows for X)
  const int j   = wv & 1;             // tile col (batch)
  const int h   = lane >> 4;
  const int col = lane & 15;

  // Hoist loop-invariant A-operand tiles into registers (6 x v16h = 48 VGPRs).
  const v16h aKm0 = load_tile16(Km, i * 16, 0,  lane);
  const v16h aKm1 = load_tile16(Km, i * 16, 32, lane);
  const v16h aCp0 = load_tile16(Cp, i * 16, 0,  lane);
  const v16h aCp1 = load_tile16(Cp, i * 16, 32, lane);
  const v16h aBp0 = load_tile16(Bp, i * 16, 0,  lane);
  const v16h aBp1 = load_tile16(Bp, i * 16, 32, lane);

  // Persistent f32 state tile: rows n = i*16 + r + 8h, cols b = j*16+col
  v8f X;
  float dec[8];
  for (int r = 0; r < 8; ++r) {
    X[r]   = x0[d * NST + i * 16 + r + 8 * h];
    dec[r] = decayf[i * 16 + r + 8 * h];
  }

  const float* ubase = u + d;
  float*       obase = out + d;

#if HAVE_TDM
  const unsigned lds_u = (unsigned)(uintptr_t)(void*)Uraw;
  if (wv == 0) tdm_load_chunk(ubase, lds_u);          // prologue: chunk 0
#endif

  for (int ch = 0; ch < NCH; ++ch) {
    const int t0 = ch * TCH;

#if HAVE_TDM
    __builtin_amdgcn_s_wait_tensorcnt(0);             // wave0's DMA done (others: no-op)
    __syncthreads();                                  // publish Uraw to all waves
    {
      // Straight-line vectorized f32 -> f16 convert: each thread owns 8
      // contiguous elements -> 2x ds_load_b128 + 1x ds_store_b128.
      const int e0 = tid * 8;
      const f32x4 a = *(const f32x4*)&Uraw[e0];
      const f32x4 b = *(const f32x4*)&Uraw[e0 + 4];
      v8h uh;
      uh[0] = (_Float16)a[0]; uh[1] = (_Float16)a[1];
      uh[2] = (_Float16)a[2]; uh[3] = (_Float16)a[3];
      uh[4] = (_Float16)b[0]; uh[5] = (_Float16)b[1];
      uh[6] = (_Float16)b[2]; uh[7] = (_Float16)b[3];
      *(v8h*)&Ut[e0] = uh;
    }
#else
    for (int e = tid; e < BATCH * TCH; e += 256) {
      const int b = e >> 6;
      const int t = e & 63;
      const size_t gi = ((size_t)b * SEQ + (size_t)(t0 + t)) * DMODEL;
      Ut[b * 64 + t] = (_Float16)ubase[gi];
      if (ch + 1 < NCH)
        __builtin_prefetch(ubase + gi + (size_t)TCH * DMODEL, 0, 1);
    }
#endif

    // Publish previous state as the f16 B-operand image (one b128 store per lane).
    {
      v8h xh;
      for (int r = 0; r < 8; ++r) xh[r] = (_Float16)X[r];
      *(v8h*)&Xt[(j * 16 + col) * 64 + i * 16 + 8 * h] = xh;
    }
    __syncthreads();                                  // Ut/Xt ready

#if HAVE_TDM
    if (wv == 0 && ch + 1 < NCH)                      // overlap next chunk's DMA
      tdm_load_chunk(ubase + (size_t)(t0 + TCH) * DMODEL, lds_u);
#endif

    const v16h bU0 = load_tile16(Ut, j * 16, 0,  lane);
    const v16h bU1 = load_tile16(Ut, j * 16, 32, lane);
    const v16h bX0 = load_tile16(Xt, j * 16, 0,  lane);
    const v16h bX1 = load_tile16(Xt, j * 16, 32, lane);

    v8f Y;
    for (int r = 0; r < 8; ++r) Y[r] = 0.0f;

    // Y = K @ U  (intra-chunk causal conv)  +  Cp @ X_prev  (carried state)
    Y = __builtin_amdgcn_wmma_f32_16x16x32_f16(false, aKm0, false, bU0, (short)0, Y, false, false);
    Y = __builtin_amdgcn_wmma_f32_16x16x32_f16(false, aKm1, false, bU1, (short)0, Y, false, false);
    Y = __builtin_amdgcn_wmma_f32_16x16x32_f16(false, aCp0, false, bX0, (short)0, Y, false, false);
    Y = __builtin_amdgcn_wmma_f32_16x16x32_f16(false, aCp1, false, bX1, (short)0, Y, false, false);

    // X = decay ⊙ X + Bp @ U   (state carried in f32 registers)
    for (int r = 0; r < 8; ++r) X[r] *= dec[r];
    X = __builtin_amdgcn_wmma_f32_16x16x32_f16(false, aBp0, false, bU0, (short)0, X, false, false);
    X = __builtin_amdgcn_wmma_f32_16x16x32_f16(false, aBp1, false, bU1, (short)0, X, false, false);

    // Store Y tile (streaming, non-temporal: keep u resident in L2):
    // rows t = t0 + i*16 + r + 8h, cols b = j*16 + col
    {
      const size_t bg = (size_t)(j * 16 + col);
      const int tb = t0 + i * 16 + 8 * h;
      for (int r = 0; r < 8; ++r)
        __builtin_nontemporal_store(Y[r], &obase[(bg * SEQ + (size_t)(tb + r)) * DMODEL]);
    }
    __syncthreads();   // protect Ut/Xt before next chunk's staging
  }
}

extern "C" void kernel_launch(void* const* d_in, const int* in_sizes, int n_in,
                              void* d_out, int out_size, void* d_ws, size_t ws_size,
                              hipStream_t stream) {
  (void)in_sizes; (void)n_in; (void)out_size; (void)d_ws; (void)ws_size;
  const float* u      = (const float*)d_in[0];
  const float* log_dt = (const float*)d_in[1];
  const float* A_real = (const float*)d_in[2];
  const float* B      = (const float*)d_in[3];
  const float* C      = (const float*)d_in[4];
  const float* x0     = (const float*)d_in[5];
  float* out = (float*)d_out;

  s5_ssd_wmma<<<dim3(DMODEL), dim3(256), 0, stream>>>(u, log_dt, A_real, B, C, x0, out);
}
